// DiffeqSolver_32933809225744
// MI455X (gfx1250) — matrix operand order repaired and checked by reading, not compiled, not run on hardware
//
#include <hip/hip_runtime.h>
#include <hip/hip_bf16.h>

typedef __attribute__((ext_vector_type(16))) __bf16 v16bf;
typedef __attribute__((ext_vector_type(8)))  __bf16 v8bf;
typedef __attribute__((ext_vector_type(8)))  float  v8f;
typedef __attribute__((ext_vector_type(4)))  int    v4i;

#if __has_builtin(__builtin_amdgcn_global_load_async_to_lds_b128)
#define USE_ASYNC_LDS 1
typedef __attribute__((address_space(1))) v4i g_v4i;  // global
typedef __attribute__((address_space(3))) v4i l_v4i;  // LDS
#define WAIT_ASYNC0() asm volatile("s_wait_asynccnt 0x0" ::: "memory")
#else
#define USE_ASYNC_LDS 0
#endif

__device__ __forceinline__ float fast_tanh(float x) {
#if __has_builtin(__builtin_amdgcn_tanhf)
    return __builtin_amdgcn_tanhf(x);   // v_tanh_f32 (CDNA5 trans op)
#else
    return tanhf(x);
#endif
}

// ---------------------------------------------------------------------------
// Prep kernels
// ---------------------------------------------------------------------------

// W: K x N (row-major fp32)  ->  Wt: N x K (row-major bf16)
__global__ void prep_transpose_bf16(const float* __restrict__ W,
                                    __bf16* __restrict__ Wt,
                                    int K, int N) {
    int i = blockIdx.x * blockDim.x + threadIdx.x;
    if (i >= K * N) return;
    int k = i / N;
    int n = i - k * N;
    Wt[(size_t)n * K + k] = (__bf16)W[i];
}

// y = first_point (fp32 state), abuf = bf16(first_point), pred[0] = first_point
__global__ void prep_init(const float* __restrict__ fp,
                          float* __restrict__ y,
                          __bf16* __restrict__ abuf,
                          float* __restrict__ out0, int n) {
    int i = blockIdx.x * blockDim.x + threadIdx.x;
    if (i >= n) return;
    float v = fp[i];
    y[i] = v;
    abuf[i] = (__bf16)v;
    out0[i] = v;
}

// ---------------------------------------------------------------------------
// Fused WMMA GEMM:  C = A[MxK](bf16) @ Bt[NxK](bf16)^T + bias[N]
// MODE 0: hid_out = bf16(tanh(C))                 (hidden layer; STAGE unused)
// MODE 1: kv = C; RK4 stage STAGE fused into epilogue (output layer)
// Block: 256 threads / 8 waves (4 x 2), block tile 128x128, K-slab 32,
// double-buffered LDS. Tiles reach LDS via GLOBAL_LOAD_ASYNC_TO_LDS_B128
// (ASYNCcnt) when available; else via named-scalar register staging.
// ---------------------------------------------------------------------------
template <int MODE, int STAGE>
__global__ __launch_bounds__(256, 1) void gemm_rk4(
    const __bf16* __restrict__ A, const __bf16* __restrict__ Bt,
    const float* __restrict__ bias, int M, int N, int K,
    __bf16* __restrict__ hid_out,
    const float* __restrict__ ts, int t,
    float* __restrict__ y, float* __restrict__ kacc,
    float* __restrict__ pred, __bf16* __restrict__ abuf) {
    constexpr int BM = 128, BN = 128, KT = 32;
    constexpr int LDP = 40;  // padded row stride (elements): 80B

    __shared__ __bf16 As[2][BM * LDP];
    __shared__ __bf16 Bs[2][BN * LDP];

    const int tid  = threadIdx.x;
    const int lane = tid & 31;
    const int w    = tid >> 5;   // wave id 0..7
    const int wm   = w >> 1;     // 0..3 : wave row  (32 rows each)
    const int wn   = w & 1;      // 0..1 : wave col  (64 cols each)
    const int l16  = lane & 15;
    const int hi   = lane >> 4;

    const int m0 = blockIdx.y * BM;
    const int n0 = blockIdx.x * BN;

    const v8f vzero = {0.f, 0.f, 0.f, 0.f, 0.f, 0.f, 0.f, 0.f};
    v8f acc[2][4];
#pragma unroll
    for (int i = 0; i < 2; ++i)
#pragma unroll
        for (int j = 0; j < 4; ++j) acc[i][j] = vzero;

    // Per-thread chunk coordinates: 128x32 tile = 512 x 16B chunks,
    // 2 chunks per thread per matrix.
    const int row0 = tid >> 2;              // 0..63
    const int col0 = (tid & 3) * 8;         // 0,8,16,24 elements
    const int row1 = row0 + 64;             // 64..127
    const int col1 = col0;

    const __bf16* gA0 = A + (size_t)(m0 + row0) * K + col0;
    const __bf16* gA1 = A + (size_t)(m0 + row1) * K + col1;
    const __bf16* gB0 = Bt + (size_t)(n0 + row0) * K + col0;
    const __bf16* gB1 = Bt + (size_t)(n0 + row1) * K + col1;
    const int la0 = row0 * LDP + col0;
    const int la1 = row1 * LDP + col1;

#if USE_ASYNC_LDS
    auto issue_async = [&](int buf, int k0) {
        __builtin_amdgcn_global_load_async_to_lds_b128(
            (g_v4i*)(void*)(gA0 + k0), (l_v4i*)(void*)&As[buf][la0], 0, 0);
        __builtin_amdgcn_global_load_async_to_lds_b128(
            (g_v4i*)(void*)(gA1 + k0), (l_v4i*)(void*)&As[buf][la1], 0, 0);
        __builtin_amdgcn_global_load_async_to_lds_b128(
            (g_v4i*)(void*)(gB0 + k0), (l_v4i*)(void*)&Bs[buf][la0], 0, 0);
        __builtin_amdgcn_global_load_async_to_lds_b128(
            (g_v4i*)(void*)(gB1 + k0), (l_v4i*)(void*)&Bs[buf][la1], 0, 0);
    };
#endif

    auto compute = [&](int cur) {
        // A fragments (16x32): lane<16 -> K {0..7,16..23}; lane>=16 -> +8
        v16bf afr[2];
#pragma unroll
        for (int mt = 0; mt < 2; ++mt) {
            const __bf16* p = &As[cur][(wm * 32 + mt * 16 + l16) * LDP];
            v8bf lo  = *(const v8bf*)(p + hi * 8);
            v8bf hi8 = *(const v8bf*)(p + 16 + hi * 8);
            afr[mt] = __builtin_shufflevector(lo, hi8, 0, 1, 2, 3, 4, 5, 6, 7,
                                              8, 9, 10, 11, 12, 13, 14, 15);
        }
        // B fragments (32x16): lane = col; lane<16 -> K {0..7,16..23}; lane>=16 -> +8
        v16bf bfr[4];
#pragma unroll
        for (int nt = 0; nt < 4; ++nt) {
            const __bf16* p =
                &Bs[cur][(wn * 64 + nt * 16 + l16) * LDP + hi * 8];
            v8bf lo  = *(const v8bf*)(p);
            v8bf hi8 = *(const v8bf*)(p + 16);
            bfr[nt] = __builtin_shufflevector(lo, hi8, 0, 1, 2, 3, 4, 5, 6, 7,
                                              8, 9, 10, 11, 12, 13, 14, 15);
        }
#pragma unroll
        for (int mt = 0; mt < 2; ++mt)
#pragma unroll
            for (int nt = 0; nt < 4; ++nt)
                acc[mt][nt] = __builtin_amdgcn_wmma_f32_16x16x32_bf16(
                    false, afr[mt], false, bfr[nt], (short)0, acc[mt][nt],
                    false, false);
    };

    const int nslab = K / KT;

#if USE_ASYNC_LDS
    issue_async(0, 0);
    WAIT_ASYNC0();
    __syncthreads();
    for (int s = 0; s < nslab; ++s) {
        const bool more = (s + 1 < nslab);
        if (more) issue_async((s + 1) & 1, (s + 1) * KT);  // DMA overlaps WMMAs
        compute(s & 1);
        if (more) WAIT_ASYNC0();                            // our 4 DMAs landed
        __syncthreads();                                    // everyone's landed
    }
#else
    // Register-staged fallback (named scalars: must stay in VGPRs, no arrays)
    uint4 ra0, ra1, rb0, rb1;
    auto load_regs = [&](int k0) {
        ra0 = *(const uint4*)(gA0 + k0);
        ra1 = *(const uint4*)(gA1 + k0);
        rb0 = *(const uint4*)(gB0 + k0);
        rb1 = *(const uint4*)(gB1 + k0);
    };
    auto store_lds = [&](int buf) {
        *(uint4*)(&As[buf][la0]) = ra0;
        *(uint4*)(&As[buf][la1]) = ra1;
        *(uint4*)(&Bs[buf][la0]) = rb0;
        *(uint4*)(&Bs[buf][la1]) = rb1;
    };
    load_regs(0);
    store_lds(0);
    __syncthreads();
    for (int s = 0; s < nslab; ++s) {
        const bool more = (s + 1 < nslab);
        if (more) load_regs((s + 1) * KT);
        compute(s & 1);
        if (more) store_lds((s + 1) & 1);
        __syncthreads();
    }
#endif

    // ------------------------------------------------------------------ epilogue
    float dt = 0.f;
    float* predT = nullptr;
    if constexpr (MODE == 1) {
        dt = ts[t + 1] - ts[t];
        predT = pred + (size_t)(t + 1) * M * N;
    }

#pragma unroll
    for (int mt = 0; mt < 2; ++mt) {
#pragma unroll
        for (int nt = 0; nt < 4; ++nt) {
            const int n     = n0 + wn * 64 + nt * 16 + l16;
            const int mbase = m0 + wm * 32 + mt * 16 + hi * 8;
            const float bv  = bias[n];
#pragma unroll
            for (int r = 0; r < 8; ++r) {
                float v = acc[mt][nt][r] + bv;
                unsigned idx = (unsigned)(mbase + r) * (unsigned)N + (unsigned)n;
                if constexpr (MODE == 0) {
                    hid_out[idx] = (__bf16)fast_tanh(v);
                } else if constexpr (STAGE == 0) {
                    kacc[idx] = v;
                    abuf[idx] = (__bf16)(y[idx] + 0.5f * dt * v);
                } else if constexpr (STAGE == 1) {
                    kacc[idx] += 2.0f * v;
                    abuf[idx] = (__bf16)(y[idx] + 0.5f * dt * v);
                } else if constexpr (STAGE == 2) {
                    kacc[idx] += 2.0f * v;
                    abuf[idx] = (__bf16)(y[idx] + dt * v);
                } else {
                    float yn = y[idx] + (dt * (1.0f / 6.0f)) * (kacc[idx] + v);
                    y[idx]     = yn;
                    predT[idx] = yn;
                    abuf[idx]  = (__bf16)yn;
                }
            }
        }
    }
}

// ---------------------------------------------------------------------------
// Host driver
// ---------------------------------------------------------------------------
extern "C" void kernel_launch(void* const* d_in, const int* in_sizes, int n_in,
                              void* d_out, int out_size, void* d_ws, size_t ws_size,
                              hipStream_t stream) {
    const float* first = (const float*)d_in[0];  // [B, D]
    const float* ts    = (const float*)d_in[1];  // [T]
    const float* W1    = (const float*)d_in[2];  // [D, H]
    const float* b1    = (const float*)d_in[3];  // [H]
    const float* W2    = (const float*)d_in[4];  // [H, D]
    const float* b2    = (const float*)d_in[5];  // [D]
    float* pred = (float*)d_out;                 // [T, B, D]

    const int T = in_sizes[1];
    const int H = in_sizes[3];
    const int D = in_sizes[5];
    const int B = in_sizes[0] / D;

    auto align256 = [](size_t x) { return (x + 255) & ~(size_t)255; };
    char* ws = (char*)d_ws;
    size_t off = 0;
    __bf16* W1t = (__bf16*)(ws + off); off = align256(off + (size_t)H * D * sizeof(__bf16)); // H x D
    __bf16* W2t = (__bf16*)(ws + off); off = align256(off + (size_t)D * H * sizeof(__bf16)); // D x H
    float*  yst = (float*) (ws + off); off = align256(off + (size_t)B * D * sizeof(float));
    float*  kac = (float*) (ws + off); off = align256(off + (size_t)B * D * sizeof(float));
    __bf16* abuf= (__bf16*)(ws + off); off = align256(off + (size_t)B * D * sizeof(__bf16));
    __bf16* hid = (__bf16*)(ws + off); off = align256(off + (size_t)B * H * sizeof(__bf16));
    (void)ws_size; (void)n_in; (void)out_size;

    prep_transpose_bf16<<<(D * H + 255) / 256, 256, 0, stream>>>(W1, W1t, D, H);
    prep_transpose_bf16<<<(H * D + 255) / 256, 256, 0, stream>>>(W2, W2t, H, D);
    prep_init<<<(B * D + 255) / 256, 256, 0, stream>>>(first, yst, abuf, pred, B * D);

    const dim3 blk(256);
    const dim3 g1(H / 128, B / 128);  // hidden GEMM: M=B, N=H, K=D
    const dim3 g2(D / 128, B / 128);  // output GEMM: M=B, N=D, K=H

    for (int t = 0; t < T - 1; ++t) {
        for (int stage = 0; stage < 4; ++stage) {
            gemm_rk4<0, 0><<<g1, blk, 0, stream>>>(abuf, W1t, b1, B, H, D, hid,
                                                   nullptr, 0, nullptr, nullptr,
                                                   nullptr, nullptr);
            switch (stage) {
            case 0:
                gemm_rk4<1, 0><<<g2, blk, 0, stream>>>(hid, W2t, b2, B, D, H,
                                                       nullptr, ts, t, yst, kac,
                                                       pred, abuf);
                break;
            case 1:
                gemm_rk4<1, 1><<<g2, blk, 0, stream>>>(hid, W2t, b2, B, D, H,
                                                       nullptr, ts, t, yst, kac,
                                                       pred, abuf);
                break;
            case 2:
                gemm_rk4<1, 2><<<g2, blk, 0, stream>>>(hid, W2t, b2, B, D, H,
                                                       nullptr, ts, t, yst, kac,
                                                       pred, abuf);
                break;
            default:
                gemm_rk4<1, 3><<<g2, blk, 0, stream>>>(hid, W2t, b2, B, D, H,
                                                       nullptr, ts, t, yst, kac,
                                                       pred, abuf);
                break;
            }
        }
    }
}
